// SoftMoE_22462678958299
// MI455X (gfx1250) — matrix operand order repaired and checked by reading, not compiled, not run on hardware
//
#include <hip/hip_runtime.h>
#include <hip/hip_bf16.h>
#include <math.h>

// ---------------- problem constants (from reference setup_inputs) ----------
#define BQ     8
#define MTOK   1024
#define DIM    768
#define NEXP   64
#define PSLOT  16
#define HID    1536
#define NP     (NEXP * PSLOT)   /* 1024 slots */

typedef __attribute__((ext_vector_type(16))) _Float16 v16h;
typedef __attribute__((ext_vector_type(8)))  _Float16 v8h;
typedef __attribute__((ext_vector_type(8)))  float    v8f;

// ===========================================================================
// 1) phi normalization over expert axis -> f16 [np][d]  (Bt layout for GEMM)
// ===========================================================================
__global__ void k_phi_norm(const float* __restrict__ phi, _Float16* __restrict__ phin) {
    int i = blockIdx.x * blockDim.x + threadIdx.x;          // over PSLOT*DIM
    if (i >= PSLOT * DIM) return;
    int pp = i / DIM, dd = i - pp * DIM;
    float s = 0.f;
    for (int e = 0; e < NEXP; ++e) {
        float v = phi[(e * PSLOT + pp) * DIM + dd];
        s += v * v;
    }
    float r = rsqrtf(s + 1e-6f);
    for (int e = 0; e < NEXP; ++e) {
        int idx = (e * PSLOT + pp) * DIM + dd;
        phin[idx] = (_Float16)(phi[idx] * r);
    }
}

// ===========================================================================
// 2) x -> f16 row-major [b*m][d]  and f16 transposed per batch [b][d][m]
// ===========================================================================
__global__ void k_cvt_x(const float* __restrict__ x,
                        _Float16* __restrict__ xb, _Float16* __restrict__ xt) {
    long i = (long)blockIdx.x * blockDim.x + threadIdx.x;
    if (i >= (long)BQ * MTOK * DIM) return;
    float v = x[i];
    xb[i] = (_Float16)v;
    int dd = (int)(i % DIM);
    long bm = i / DIM;
    int mm = (int)(bm % MTOK);
    int b  = (int)(bm / MTOK);
    xt[((long)b * DIM + dd) * MTOK + mm] = (_Float16)v;
}

// ===========================================================================
// 3) weight transpose+convert: w[n][R][C] f32 -> wt[n][C][R] f16
// ===========================================================================
__global__ void k_cvt_T(const float* __restrict__ w, _Float16* __restrict__ wt,
                        int R, int C) {
    long base = (long)blockIdx.y * R * C;
    long tot  = (long)R * C;
    for (long i = (long)blockIdx.x * blockDim.x + threadIdx.x; i < tot;
         i += (long)gridDim.x * blockDim.x) {
        int rr = (int)(i / C), cc = (int)(i - (long)rr * C);
        wt[base + (long)cc * R + rr] = (_Float16)w[base + i];
    }
}

// ===========================================================================
// 4) generic TN WMMA GEMM, 4x4 register-tiled: each wave computes a 64x64
//    output block (16 WMMA tiles) -> 16 B128 loads feed 16 WMMAs per K-step.
//    C[M][N] f32 = A[M][K] f16 * Bt[N][K] f16 ; M,N multiples of 64,
//    K multiple of 32.
// ===========================================================================
__global__ __launch_bounds__(256) void k_gemm_tn(
    const _Float16* __restrict__ A, const _Float16* __restrict__ Bt,
    float* __restrict__ Cout, int Mm, int Nn, int Kk,
    long strideA, long strideB, long strideC)
{
    int tilesN = Nn >> 6;                       // 64-wide blocks
    int total  = (Mm >> 6) * tilesN;
    int wave   = blockIdx.x * 8 + (threadIdx.x >> 5);
    if (wave >= total) return;
    int lane = threadIdx.x & 31;
    int tm = wave / tilesN, tn = wave - tm * tilesN;
    int lo = lane & 15, hi = lane >> 4;
    long bo = blockIdx.y;

    // per-lane row pointers: A row M, per-lane K runs [8*hi,8*hi+8) and +16;
    // B row N, per-lane K runs [8*hi,8*hi+8) and +16
    const _Float16* Ap[4];
    const _Float16* Bp[4];
#pragma unroll
    for (int i = 0; i < 4; ++i) {
        Ap[i] = A  + bo * strideA + ((long)(tm * 64 + i * 16 + lo)) * Kk + hi * 8;
        Bp[i] = Bt + bo * strideB + ((long)(tn * 64 + i * 16 + lo)) * Kk + hi * 8;
    }

    v8f acc[16];
#pragma unroll
    for (int i = 0; i < 16; ++i) acc[i] = (v8f){};

    for (int k0 = 0; k0 < Kk; k0 += 32) {
        union { v16h v; v8h h[2]; } a[4], bmat[4];
#pragma unroll
        for (int i = 0; i < 4; ++i) {
            a[i].h[0]    = *(const v8h*)(Ap[i] + k0);
            a[i].h[1]    = *(const v8h*)(Ap[i] + k0 + 16);
            bmat[i].h[0] = *(const v8h*)(Bp[i] + k0);
            bmat[i].h[1] = *(const v8h*)(Bp[i] + k0 + 16);
        }
        // prefetch next K chunk of the streamed rows (global_prefetch_b8)
        if (k0 + 128 < Kk) {
            __builtin_prefetch(Ap[0] + k0 + 128, 0, 1);
            __builtin_prefetch(Bp[0] + k0 + 128, 0, 1);
        }
#pragma unroll
        for (int i = 0; i < 4; ++i)
#pragma unroll
            for (int j = 0; j < 4; ++j)
                acc[i * 4 + j] = __builtin_amdgcn_wmma_f32_16x16x32_f16(
                    false, a[i].v, false, bmat[j].v, (short)0, acc[i * 4 + j],
                    false, false);
    }

    // C/D layout: lane -> col N = lo, rows M = r + 8*hi
#pragma unroll
    for (int i = 0; i < 4; ++i) {
#pragma unroll
        for (int j = 0; j < 4; ++j) {
            float* Cp = Cout + bo * strideC +
                        ((long)(tm * 64 + i * 16 + 8 * hi)) * Nn +
                        tn * 64 + j * 16 + lo;
#pragma unroll
            for (int r = 0; r < 8; ++r) Cp[(long)r * Nn] = acc[i * 4 + j][r];
        }
    }
}

// ===========================================================================
// 5a) softmax over slots (contiguous): C[b][m][np] f16
// ===========================================================================
__global__ __launch_bounds__(256) void k_softmax_slots(
    const float* __restrict__ logits, _Float16* __restrict__ Cm)
{
    __shared__ float red[8];
    long row = blockIdx.x;                       // b*MTOK + m
    const float* src = logits + row * NP;
    int t = threadIdx.x, lane = t & 31, wv = t >> 5;
    float v[4], mx = -1e30f;
#pragma unroll
    for (int j = 0; j < 4; ++j) { v[j] = src[t + j * 256]; mx = fmaxf(mx, v[j]); }
    for (int o = 16; o > 0; o >>= 1) mx = fmaxf(mx, __shfl_xor(mx, o, 32));
    if (lane == 0) red[wv] = mx;
    __syncthreads();
    float m2 = red[0];
#pragma unroll
    for (int i = 1; i < 8; ++i) m2 = fmaxf(m2, red[i]);
    float sum = 0.f;
#pragma unroll
    for (int j = 0; j < 4; ++j) { v[j] = expf(v[j] - m2); sum += v[j]; }
    for (int o = 16; o > 0; o >>= 1) sum += __shfl_xor(sum, o, 32);
    __syncthreads();
    if (lane == 0) red[wv] = sum;
    __syncthreads();
    float tot = 0.f;
#pragma unroll
    for (int i = 0; i < 8; ++i) tot += red[i];
    float inv = 1.f / tot;
    _Float16* dst = Cm + row * NP;
#pragma unroll
    for (int j = 0; j < 4; ++j) dst[t + j * 256] = (_Float16)(v[j] * inv);
}

// ===========================================================================
// 5b) softmax over tokens (strided read), transposed write: Dt[b][slot][m] f16
// ===========================================================================
__global__ __launch_bounds__(256) void k_softmax_tokens(
    const float* __restrict__ logits, _Float16* __restrict__ Dt)
{
    __shared__ float red[8];
    int b = blockIdx.x / NP, s = blockIdx.x - b * NP;
    const float* col = logits + (long)b * MTOK * NP + s;
    int t = threadIdx.x, lane = t & 31, wv = t >> 5;
    float v[4], mx = -1e30f;
#pragma unroll
    for (int j = 0; j < 4; ++j) { v[j] = col[(long)(t + j * 256) * NP]; mx = fmaxf(mx, v[j]); }
    for (int o = 16; o > 0; o >>= 1) mx = fmaxf(mx, __shfl_xor(mx, o, 32));
    if (lane == 0) red[wv] = mx;
    __syncthreads();
    float m2 = red[0];
#pragma unroll
    for (int i = 1; i < 8; ++i) m2 = fmaxf(m2, red[i]);
    float sum = 0.f;
#pragma unroll
    for (int j = 0; j < 4; ++j) { v[j] = expf(v[j] - m2); sum += v[j]; }
    for (int o = 16; o > 0; o >>= 1) sum += __shfl_xor(sum, o, 32);
    __syncthreads();
    if (lane == 0) red[wv] = sum;
    __syncthreads();
    float tot = 0.f;
#pragma unroll
    for (int i = 0; i < 8; ++i) tot += red[i];
    float inv = 1.f / tot;
    _Float16* dst = Dt + ((long)b * NP + s) * MTOK;
#pragma unroll
    for (int j = 0; j < 4; ++j) dst[t + j * 256] = (_Float16)(v[j] * inv);
}

// ===========================================================================
// 6) fused per-(batch,expert) MLP: LN -> W1+b1 -> GELU -> W2+b2,
//    writes Ys transposed as f16: Yst[b][d][np]
//    LDS: sA (16x768 f16, 24KB) + sH (16x768 f16, 24KB) = 48KB static
// ===========================================================================
__global__ __launch_bounds__(256) void k_expert_mlp(
    const float* __restrict__ Xs, const float* __restrict__ ln_g,
    const float* __restrict__ ln_b, const _Float16* __restrict__ w1t,
    const float* __restrict__ b1, const _Float16* __restrict__ w2t,
    const float* __restrict__ b2, _Float16* __restrict__ Yst)
{
    __shared__ _Float16 sA[16 * DIM];   // LayerNormed slot inputs
    __shared__ _Float16 sH[16 * DIM];   // half of GELU(hdn), 768 cols at a time

    int b = blockIdx.x / NEXP, n = blockIdx.x - b * NEXP;
    int t = threadIdx.x, lane = t & 31, wv = t >> 5;
    int lo = lane & 15, hi = lane >> 4;

    const float* Xb = Xs + ((long)b * NP + n * PSLOT) * DIM;   // [16][768]
    // ---- LayerNorm: 8 waves x 2 rows -------------------------------------
    for (int rr = wv * 2; rr < wv * 2 + 2; ++rr) {
        const float* row = Xb + rr * DIM;
        float s = 0.f, s2 = 0.f;
        for (int c = lane; c < DIM; c += 32) { float v = row[c]; s += v; s2 += v * v; }
        for (int o = 16; o > 0; o >>= 1) { s += __shfl_xor(s, o, 32); s2 += __shfl_xor(s2, o, 32); }
        float mu   = s * (1.0f / DIM);
        float var  = s2 * (1.0f / DIM) - mu * mu;
        float rstd = rsqrtf(var + 1e-5f);
        for (int c = lane; c < DIM; c += 32) {
            float v = (row[c] - mu) * rstd * ln_g[n * DIM + c] + ln_b[n * DIM + c];
            sA[rr * DIM + c] = (_Float16)v;
        }
    }
    __syncthreads();

    const _Float16* W1 = w1t + (long)n * HID * DIM;   // [h][d]
    const _Float16* W2 = w2t + (long)n * DIM * HID;   // [d][h]

    v8f acc2[6];                                       // 6 output d-tiles/wave
#pragma unroll
    for (int i = 0; i < 6; ++i) acc2[i] = (v8f){};

    for (int half = 0; half < 2; ++half) {
        // ---- GEMM1 (+bias+GELU) for hdn cols [half*768, half*768+768) ----
        for (int tc = wv; tc < DIM / 16; tc += 8) {
            int colg = half * DIM + tc * 16 + lo;       // global h index
            v8f acc = {};
            const _Float16* bp = W1 + (long)colg * DIM + hi * 8;
            const _Float16* ap = sA + lo * DIM + hi * 8;
#pragma unroll 4
            for (int k0 = 0; k0 < DIM; k0 += 32) {
                union { v16h v; v8h h[2]; } a, bmat;
                a.h[0]    = *(const v8h*)(ap + k0);
                a.h[1]    = *(const v8h*)(ap + k0 + 16);
                bmat.h[0] = *(const v8h*)(bp + k0);
                bmat.h[1] = *(const v8h*)(bp + k0 + 16);
                acc = __builtin_amdgcn_wmma_f32_16x16x32_f16(
                    false, a.v, false, bmat.v, (short)0, acc, false, false);
            }
            float bias = b1[n * HID + colg];
            int lc = tc * 16 + lo;                      // local col in sH
#pragma unroll
            for (int r = 0; r < 8; ++r) {
                float v = acc[r] + bias;
                v = 0.5f * v * (1.0f + erff(v * 0.70710678118654752f)); // exact GELU
                sH[(r + 8 * hi) * DIM + lc] = (_Float16)v;
            }
        }
        __syncthreads();
        // ---- GEMM2 partial accumulation over this K half -----------------
#pragma unroll
        for (int tt = 0; tt < 6; ++tt) {
            int tc = wv + tt * 8;                        // output d tile
            const _Float16* bp = W2 + (long)(tc * 16 + lo) * HID + half * DIM + hi * 8;
            const _Float16* ap = sH + lo * DIM + hi * 8;
            v8f acc = acc2[tt];
#pragma unroll 4
            for (int k0 = 0; k0 < DIM; k0 += 32) {
                union { v16h v; v8h h[2]; } a, bmat;
                a.h[0]    = *(const v8h*)(ap + k0);
                a.h[1]    = *(const v8h*)(ap + k0 + 16);
                bmat.h[0] = *(const v8h*)(bp + k0);
                bmat.h[1] = *(const v8h*)(bp + k0 + 16);
                acc = __builtin_amdgcn_wmma_f32_16x16x32_f16(
                    false, a.v, false, bmat.v, (short)0, acc, false, false);
            }
            acc2[tt] = acc;
        }
        __syncthreads();   // sH reused next half
    }

    // ---- epilogue: +b2, write Yst[b][d][slot] (contiguous 8 f16 per lane) -
#pragma unroll
    for (int tt = 0; tt < 6; ++tt) {
        int tc  = wv + tt * 8;
        int col = tc * 16 + lo;                          // d index
        float bias = b2[n * DIM + col];
        v8h out;
#pragma unroll
        for (int r = 0; r < 8; ++r) out[r] = (_Float16)(acc2[tt][r] + bias);
        _Float16* yp = Yst + ((long)b * DIM + col) * NP + n * PSLOT + 8 * hi;
        *(v8h*)yp = out;
    }
}

// ===========================================================================
// host-side launcher
// ===========================================================================
static inline size_t alignup(size_t x) { return (x + 255) & ~(size_t)255; }

extern "C" void kernel_launch(void* const* d_in, const int* in_sizes, int n_in,
                              void* d_out, int out_size, void* d_ws, size_t ws_size,
                              hipStream_t stream) {
    const float* x    = (const float*)d_in[0];
    const float* phi  = (const float*)d_in[1];
    const float* ln_g = (const float*)d_in[2];
    const float* ln_b = (const float*)d_in[3];
    const float* w1   = (const float*)d_in[4];
    const float* b1   = (const float*)d_in[5];
    const float* w2   = (const float*)d_in[6];
    const float* b2   = (const float*)d_in[7];
    float* out = (float*)d_out;

    // ---- workspace layout ------------------------------------------------
    char* w = (char*)d_ws;
    size_t off = 0;
    _Float16* xb   = (_Float16*)(w + off); off = alignup(off + (size_t)BQ * MTOK * DIM * 2);
    _Float16* xt   = (_Float16*)(w + off); off = alignup(off + (size_t)BQ * DIM * MTOK * 2);
    _Float16* phin = (_Float16*)(w + off); off = alignup(off + (size_t)NP * DIM * 2);
    _Float16* w1t  = (_Float16*)(w + off); off = alignup(off + (size_t)NEXP * HID * DIM * 2);
    _Float16* w2t  = (_Float16*)(w + off); off = alignup(off + (size_t)NEXP * DIM * HID * 2);
    float*    lg   = (float*)   (w + off); off = alignup(off + (size_t)BQ * MTOK * NP * 4);
    _Float16* Dt   = (_Float16*)(w + off); off = alignup(off + (size_t)BQ * NP * MTOK * 2);
    _Float16* Cm   = (_Float16*)(w + off); off = alignup(off + (size_t)BQ * MTOK * NP * 2);
    float*    Xs   = (float*)   (w + off); off = alignup(off + (size_t)BQ * NP * DIM * 4);
    _Float16* Yst  = (_Float16*)(w + off); off = alignup(off + (size_t)BQ * DIM * NP * 2);
    (void)ws_size; (void)in_sizes; (void)n_in; (void)out_size;

    // ---- 1: phi normalization -------------------------------------------
    k_phi_norm<<<(PSLOT * DIM + 255) / 256, 256, 0, stream>>>(phi, phin);
    // ---- 2: x conversions -----------------------------------------------
    {
        long tot = (long)BQ * MTOK * DIM;
        k_cvt_x<<<(unsigned)((tot + 255) / 256), 256, 0, stream>>>(x, xb, xt);
    }
    // ---- 3: weight transposes -------------------------------------------
    k_cvt_T<<<dim3(1152, NEXP), 256, 0, stream>>>(w1, w1t, DIM, HID);
    k_cvt_T<<<dim3(1152, NEXP), 256, 0, stream>>>(w2, w2t, HID, DIM);
    // ---- 4: logits = x . phi_n  [8192x1024, K=768] ----------------------
    {
        int tiles = (BQ * MTOK / 64) * (NP / 64);      // 2048 waves
        k_gemm_tn<<<dim3((tiles + 7) / 8, 1), 256, 0, stream>>>(
            xb, phin, lg, BQ * MTOK, NP, DIM, 0L, 0L, 0L);
    }
    // ---- 5: softmaxes ----------------------------------------------------
    k_softmax_slots<<<BQ * MTOK, 256, 0, stream>>>(lg, Cm);
    k_softmax_tokens<<<BQ * NP, 256, 0, stream>>>(lg, Dt);
    // ---- 6: Xs = D^T . x per batch  [1024x768, K=1024] ------------------
    {
        int tiles = (NP / 64) * (DIM / 64);            // 192 waves/batch
        k_gemm_tn<<<dim3((tiles + 7) / 8, BQ), 256, 0, stream>>>(
            Dt, xt, Xs, NP, DIM, MTOK,
            (long)NP * MTOK, (long)DIM * MTOK, (long)NP * DIM);
    }
    // ---- 7: fused expert MLP --------------------------------------------
    k_expert_mlp<<<BQ * NEXP, 256, 0, stream>>>(Xs, ln_g, ln_b, w1t, b1, w2t, b2, Yst);
    // ---- 8: Y = C . Ys per batch  [1024x768, K=1024] --------------------
    {
        int tiles = (MTOK / 64) * (DIM / 64);          // 192 waves/batch
        k_gemm_tn<<<dim3((tiles + 7) / 8, BQ), 256, 0, stream>>>(
            Cm, Yst, out, MTOK, DIM, MTOK,
            (long)MTOK * NP, (long)DIM * NP, (long)MTOK * DIM);
    }
}
